// RPN_23012434772604
// MI455X (gfx1250) — matrix operand order repaired and checked by reading, not compiled, not run on hardware
//
#include <hip/hip_runtime.h>

#define B_       8
#define CIN      512
#define COUT     512
#define HH       64
#define WW       64
#define KTOT     (9 * CIN)        // 4608
#define NPIX     (B_ * HH * WW)   // 32768
#define APB      9
#define NCAT     64               // 54 valid, padded
#define NANCH    (HH * WW * APB)  // 36864
#define PRE_NMS  6000
#define POST_NMS 300
#define NMS_TH   0.7f
#define MIN_SIZE 16.0f
#define NWORDS   188              // ceil(6000/32)

#define LDK      40               // LDS row stride in halves (80B, 16B aligned)
#define AROW     (66 * LDK)       // halo'd A buffer: x' = 0..65 (x = -1..64)

typedef __attribute__((ext_vector_type(16))) _Float16 v16h;
typedef __attribute__((ext_vector_type(8)))  _Float16 v8h;
typedef __attribute__((ext_vector_type(4)))  _Float16 v4h;
typedef __attribute__((ext_vector_type(8)))  float    v8f;
typedef __attribute__((ext_vector_type(4)))  float    v4f;

struct HPair { v8h lo; v8h hi; };
union  HF16  { v16h v; HPair s; };

// A-matrix fragment (16x32 f16), ISA 7.12.2 layout:
// lane L holds row M=L%16; lanes 0-15 hold K {0..7, 16..23}, lanes 16-31 hold K {8..15, 24..31}
__device__ __forceinline__ v16h frag_a(const _Float16* base, int lane, int ldk) {
    const _Float16* row = base + (lane & 15) * ldk;
    const int k0 = (lane < 16) ? 0 : 8;
    HF16 u;
    u.s.lo = *(const v8h*)(row + k0);
    u.s.hi = *(const v8h*)(row + k0 + 16);
    return u.v;
}

// B-matrix fragment (32x16 f16): lane L holds col N=L%16; lanes 0-15 K {0..7, 16..23}, lanes 16-31 K {8..15, 24..31}
__device__ __forceinline__ v16h frag_b(const _Float16* base, int lane, int ldk) {
    const _Float16* row = base + (lane & 15) * ldk;
    const int k0 = (lane < 16) ? 0 : 8;
    HF16 u;
    u.s.lo = *(const v8h*)(row + k0);
    u.s.hi = *(const v8h*)(row + k0 + 16);
    return u.v;
}

__device__ __forceinline__ unsigned f2key(float f) {
    unsigned u = __float_as_uint(f);
    return (u & 0x80000000u) ? ~u : (u | 0x80000000u);
}

// ---------------- input f32 -> f16 (one time) ----------------
__global__ __launch_bounds__(256) void pack_in16(const float* __restrict__ in,
                                                 _Float16* __restrict__ o) {
    int i = blockIdx.x * 256 + threadIdx.x;      // over NPIX*CIN/4
    v4f v = *(const v4f*)(in + (size_t)i * 4);
    v4h h;
    h[0] = (_Float16)v[0]; h[1] = (_Float16)v[1];
    h[2] = (_Float16)v[2]; h[3] = (_Float16)v[3];
    *(v4h*)(o + (size_t)i * 4) = h;
}

// ---------------- weight packing: wf2[kchunk][n][kk] ----------------
__global__ __launch_bounds__(256) void pack_wfeat2(const float* __restrict__ w,
                                                   _Float16* __restrict__ wf2) {
    int idx = blockIdx.x * 256 + threadIdx.x;    // over KTOT*COUT
    if (idx >= KTOT * COUT) return;
    int kk  = idx & 31;
    int n   = (idx >> 5) & 511;
    int kc  = idx >> 14;
    int k   = kc * 32 + kk;                      // tap*512 + cin
    int tap = k >> 9;
    int cin = k & 511;
    wf2[idx] = (_Float16)w[(n * CIN + cin) * 9 + tap];
}

__global__ __launch_bounds__(256) void pack_wcat2(const float* __restrict__ wc,
                                                  const float* __restrict__ wr,
                                                  const float* __restrict__ bc,
                                                  const float* __restrict__ br,
                                                  _Float16* __restrict__ wcat2,
                                                  float* __restrict__ bcat) {
    int idx = blockIdx.x * 256 + threadIdx.x;    // over 512*64
    if (idx < 512 * NCAT) {
        int kk = idx & 31;
        int n  = (idx >> 5) & 63;
        int kc = idx >> 11;
        int k  = kc * 32 + kk;
        float v = 0.f;
        if (n < 18)      v = wc[n * 512 + k];
        else if (n < 54) v = wr[(n - 18) * 512 + k];
        wcat2[idx] = (_Float16)v;
    }
    if (blockIdx.x == 0 && threadIdx.x < NCAT) {
        int n = threadIdx.x;
        float v = 0.f;
        if (n < 18)      v = bc[n];
        else if (n < 54) v = br[n - 18];
        bcat[n] = v;
    }
}

// ---------------- 3x3 conv + ReLU as implicit GEMM (WMMA f16) ----------------
// M-tile 128 (two image rows), N-tile 128, K-step 32.
// A is staged once per (ty, c-chunk) into a halo'd buffer [x'=-1..64][c];
// the dx in {-1,0,1} taps reuse it via a whole-row base offset (stays 16B aligned).
__global__ __launch_bounds__(256)
void conv3x3_relu_wmma(const _Float16* __restrict__ in16, const _Float16* __restrict__ wf2,
                       const float* __restrict__ bias, _Float16* __restrict__ feat) {
    __shared__ _Float16 sAh[2 * AROW];        // two output rows
    __shared__ _Float16 sB[128 * LDK];
    const int tid  = threadIdx.x;
    const int lane = tid & 31;
    const int wv   = tid >> 5;
    const int q    = wv & 3;            // M-subtile within a row (x/16)
    const int nh   = wv >> 2;           // 0..1 -> n-offset 64*nh
    const int mBase = blockIdx.x * 128; // two (b,y) image rows
    const int nBase = blockIdx.y * 128;
    const int b  = mBase >> 12;
    const int y0 = (mBase >> 6) & 63;   // first output row (even)

    v8f acc[2][4] = {{{}, {}, {}, {}}, {{}, {}, {}, {}}};

    // staging roles
    const int ac  = tid >> 3;           // cin 0..31
    const int ax0 = (tid & 7) * 8;      // x chunk base
    const int bn  = tid & 127;          // B n row
    const int bh  = tid >> 7;           // B k half (0..1 -> kk 0/16)

    // zero the x=-1 / x=64 halo rows once (never overwritten)
    if (tid < 128) {
        int r = tid >> 6;               // buffer
        int c = tid & 31;
        int e = (tid >> 5) & 1;         // 0 -> x'=0, 1 -> x'=65
        sAh[r * AROW + (e ? 65 : 0) * LDK + c] = (_Float16)0.f;
    }

    for (int ty = 0; ty < 3; ++ty) {
        for (int cc = 0; cc < 16; ++cc) {
            const int c0 = cc * 32;
            for (int dxi = 0; dxi < 3; ++dxi) {
                __syncthreads();        // prior readers of sAh/sB are done
                if (dxi == 0) {
                    // ---- stage A halo buffers for both output rows ----
                    #pragma unroll
                    for (int r = 0; r < 2; ++r) {
                        const int yy = y0 + r + ty - 1;
                        _Float16* dst = sAh + r * AROW;
                        if ((unsigned)yy < 64u) {
                            const _Float16* src =
                                in16 + (((size_t)(b * CIN + c0 + ac) * 64 + yy) << 6) + ax0;
                            v8h v = *(const v8h*)src;
                            #pragma unroll
                            for (int j = 0; j < 8; ++j)
                                dst[(ax0 + 1 + j) * LDK + ac] = v[j];
                        } else {
                            #pragma unroll
                            for (int j = 0; j < 8; ++j)
                                dst[(ax0 + 1 + j) * LDK + ac] = (_Float16)0.f;
                        }
                    }
                }
                // ---- stage B: pre-swizzled, pure vector copy ----
                {
                    const int kc2 = (ty * 3 + dxi) * 16 + cc;
                    const _Float16* src = wf2 + ((size_t)kc2 * COUT + nBase + bn) * 32 + bh * 16;
                    v8h v0 = *(const v8h*)src;
                    v8h v1 = *(const v8h*)(src + 8);
                    *(v8h*)(sB + bn * LDK + bh * 16)     = v0;
                    *(v8h*)(sB + bn * LDK + bh * 16 + 8) = v1;
                    if (dxi < 2)
                        __builtin_prefetch(src + (size_t)16 * COUT * 32, 0, 1);
                }
                __syncthreads();
                // ---- compute: 2 A frags (one per row), 4 B frags, 8 WMMA ----
                const int xoff = (q * 16 + dxi) * LDK;   // (1+dx) row shift, dx = dxi-1
                v16h a0 = frag_a(sAh + 0 * AROW + xoff, lane, LDK);
                v16h a1 = frag_a(sAh + 1 * AROW + xoff, lane, LDK);
                #pragma unroll
                for (int j = 0; j < 4; ++j) {
                    v16h bf = frag_b(sB + (nh * 64 + j * 16) * LDK, lane, LDK);
                    acc[0][j] = __builtin_amdgcn_wmma_f32_16x16x32_f16(false, a0, false, bf,
                                                                       (short)0, acc[0][j], false, false);
                    acc[1][j] = __builtin_amdgcn_wmma_f32_16x16x32_f16(false, a1, false, bf,
                                                                       (short)0, acc[1][j], false, false);
                }
            }
        }
    }
    // epilogue: bias + ReLU, store f16 feat[pixel][cout]
    const int hi8 = (lane >> 4) << 3;
    const int nl  = lane & 15;
    #pragma unroll
    for (int j = 0; j < 4; ++j) {
        int gn = nBase + nh * 64 + j * 16 + nl;
        float bz = bias[gn];
        #pragma unroll
        for (int r = 0; r < 2; ++r) {
            #pragma unroll
            for (int rr = 0; rr < 8; ++rr) {
                int P = mBase + (r * 4 + q) * 16 + rr + hi8;
                float v = acc[r][j][rr] + bz;
                v = v > 0.f ? v : 0.f;
                feat[(size_t)P * COUT + gn] = (_Float16)v;
            }
        }
    }
}

// ---------------- cls/reg 1x1 heads (WMMA) + softmax + box decode ----------------
__global__ __launch_bounds__(256)
void head_decode_wmma(const _Float16* __restrict__ feat, const _Float16* __restrict__ wcat2,
                      const float* __restrict__ bcat, const float* __restrict__ im_info,
                      float* __restrict__ scores, float* __restrict__ boxes) {
    __shared__ _Float16 sA[64 * LDK];
    __shared__ _Float16 sB[64 * LDK];
    __shared__ float    sOut[64 * 66];
    const int tid  = threadIdx.x;
    const int lane = tid & 31;
    const int wv   = tid >> 5;
    const int mi   = wv & 3;
    const int ni   = wv >> 2;
    const int mBase = blockIdx.x * 64;

    v8f acc0 = {}; v8f acc1 = {};
    for (int kBase = 0; kBase < CIN; kBase += 32) {
        {   // A: one v8h per thread
            int m  = tid >> 2;
            int ko = (tid & 3) * 8;
            *(v8h*)(sA + m * LDK + ko) =
                *(const v8h*)(feat + (size_t)(mBase + m) * COUT + kBase + ko);
        }
        {   // B: vector copy from pre-swizzled wcat2[kc][n][32]
            int n = tid >> 2;
            int qq = (tid & 3) * 8;
            *(v8h*)(sB + n * LDK + qq) =
                *(const v8h*)(wcat2 + ((size_t)(kBase >> 5) * NCAT + n) * 32 + qq);
        }
        __syncthreads();
        v16h a  = frag_a(sA + mi * 16 * LDK, lane, LDK);
        v16h b0 = frag_b(sB + (ni * 32) * LDK, lane, LDK);
        v16h b1 = frag_b(sB + (ni * 32 + 16) * LDK, lane, LDK);
        acc0 = __builtin_amdgcn_wmma_f32_16x16x32_f16(false, a, false, b0, (short)0, acc0, false, false);
        acc1 = __builtin_amdgcn_wmma_f32_16x16x32_f16(false, a, false, b1, (short)0, acc1, false, false);
        __syncthreads();
    }
    const int hi8 = (lane >> 4) << 3;
    const int nl  = lane & 15;
    float bz0 = bcat[ni * 32 + nl];
    float bz1 = bcat[ni * 32 + 16 + nl];
    #pragma unroll
    for (int r = 0; r < 8; ++r) {
        int m = mi * 16 + r + hi8;
        sOut[m * 66 + ni * 32 + nl]      = acc0[r] + bz0;
        sOut[m * 66 + ni * 32 + 16 + nl] = acc1[r] + bz1;
    }
    __syncthreads();

    const int b = mBase >> 12;
    const int y = (mBase >> 6) & 63;
    const float imh  = im_info[b * 3 + 0];
    const float imw  = im_info[b * 3 + 1];
    const float imsc = im_info[b * 3 + 2];
    for (int t = tid; t < 64 * APB; t += 256) {
        int p  = t / APB;
        int a9 = t % APB;
        int x  = p;
        float c0v = sOut[p * 66 + 2 * a9];
        float c1v = sOut[p * 66 + 2 * a9 + 1];
        float fg  = 1.f / (1.f + __expf(c0v - c1v));
        float d0 = sOut[p * 66 + 18 + 4 * a9 + 0];
        float d1 = sOut[p * 66 + 18 + 4 * a9 + 1];
        float d2 = sOut[p * 66 + 18 + 4 * a9 + 2];
        float d3 = sOut[p * 66 + 18 + 4 * a9 + 3];
        const float wsT[3] = {23.f, 16.f, 11.f};
        const float hsT[3] = {12.f, 16.f, 22.f};
        const float scT[3] = {8.f, 16.f, 32.f};
        float aw = wsT[a9 / 3] * scT[a9 % 3];
        float ah = hsT[a9 / 3] * scT[a9 % 3];
        float acx = (float)x * 16.f + 8.f;
        float acy = (float)y * 16.f + 8.f;
        float pcx = d0 * aw + acx;
        float pcy = d1 * ah + acy;
        float pw  = __expf(d2) * aw;
        float ph  = __expf(d3) * ah;
        float x1 = fminf(fmaxf(pcx - 0.5f * pw, 0.f), imw - 1.f);
        float y1 = fminf(fmaxf(pcy - 0.5f * ph, 0.f), imh - 1.f);
        float x2 = fminf(fmaxf(pcx + 0.5f * pw, 0.f), imw - 1.f);
        float y2 = fminf(fmaxf(pcy + 0.5f * ph, 0.f), imh - 1.f);
        float bw = x2 - x1 + 1.f, bh = y2 - y1 + 1.f;
        float msz = MIN_SIZE * imsc;
        if (!(bw >= msz && bh >= msz)) fg = -1e9f;
        int idx = (y * WW + x) * APB + a9;
        scores[(size_t)b * NANCH + idx] = fg;
        float* bo = boxes + ((size_t)b * NANCH + idx) * 4;
        bo[0] = x1; bo[1] = y1; bo[2] = x2; bo[3] = y2;
    }
}

// ---------------- top-6000 selection via bucket histogram ----------------
__global__ __launch_bounds__(1024)
void topk_select(const float* __restrict__ scores, const float* __restrict__ boxes,
                 float* __restrict__ selS, float* __restrict__ selB) {
    __shared__ unsigned hist[2048];
    __shared__ int s_tb;
    const int b   = blockIdx.x;
    const int tid = threadIdx.x;
    for (int i = tid; i < 2048; i += 1024) hist[i] = 0;
    __syncthreads();
    const float* sc = scores + (size_t)b * NANCH;
    for (int i = tid; i < NANCH; i += 1024)
        atomicAdd(&hist[f2key(sc[i]) >> 21], 1u);
    __syncthreads();
    if (tid == 0) {
        unsigned run = 0;
        int tb = 0;
        for (int bk = 2047; bk >= 0; --bk) {
            unsigned c = hist[bk];
            hist[bk] = run;          // becomes start offset
            run += c;
            if (run >= PRE_NMS) { tb = bk; break; }
        }
        s_tb = tb;
    }
    __syncthreads();
    const int tb = s_tb;
    for (int i = tid; i < NANCH; i += 1024) {
        float s = sc[i];
        int bk = (int)(f2key(s) >> 21);
        if (bk >= tb) {
            unsigned pos = atomicAdd(&hist[bk], 1u);
            if (pos < PRE_NMS) {
                selS[(size_t)b * PRE_NMS + pos] = s;
                const float* bi = boxes + ((size_t)b * NANCH + i) * 4;
                float* bo = selB + ((size_t)b * PRE_NMS + pos) * 4;
                bo[0] = bi[0]; bo[1] = bi[1]; bo[2] = bi[2]; bo[3] = bi[3];
            }
        }
    }
}

// ---------------- NMS (300 fixed iterations) + roi emit ----------------
__global__ __launch_bounds__(256)
void nms_kernel(const float* __restrict__ selS, const float* __restrict__ selB,
                float* __restrict__ out) {
    __shared__ unsigned valid[NWORDS];
    __shared__ int keep[POST_NMS];
    __shared__ int s_word;
    __shared__ int s_idx;
    __shared__ float cb[4];
    const int b   = blockIdx.x;
    const int tid = threadIdx.x;
    const float* S  = selS + (size_t)b * PRE_NMS;
    const float* Bx = selB + (size_t)b * PRE_NMS * 4;
    for (int w = tid; w < NWORDS; w += 256) {
        unsigned m = 0;
        for (int bit = 0; bit < 32; ++bit) {
            int i = w * 32 + bit;
            if (i < PRE_NMS && S[i] > -1e8f) m |= (1u << bit);
        }
        valid[w] = m;
    }
    __syncthreads();
    for (int it = 0; it < POST_NMS; ++it) {
        if (tid == 0) s_word = NWORDS;
        __syncthreads();
        if (tid < NWORDS && valid[tid]) atomicMin(&s_word, tid);
        __syncthreads();
        if (tid == 0) {
            int wsel = s_word;
            if (wsel < NWORDS) {
                int bit = __ffs(valid[wsel]) - 1;
                int idx = wsel * 32 + bit;
                s_idx = idx;
                keep[it] = idx;
                cb[0] = Bx[idx * 4 + 0]; cb[1] = Bx[idx * 4 + 1];
                cb[2] = Bx[idx * 4 + 2]; cb[3] = Bx[idx * 4 + 3];
            } else { s_idx = -1; keep[it] = -1; }
        }
        __syncthreads();
        const int idx = s_idx;
        if (idx >= 0) {
            const float X1 = cb[0], Y1 = cb[1], X2 = cb[2], Y2 = cb[3];
            const float areaI = (X2 - X1 + 1.f) * (Y2 - Y1 + 1.f);
            for (int j = tid; j < PRE_NMS; j += 256) {
                if (!(valid[j >> 5] & (1u << (j & 31)))) continue;
                float x1 = Bx[j * 4 + 0], y1 = Bx[j * 4 + 1];
                float x2 = Bx[j * 4 + 2], y2 = Bx[j * 4 + 3];
                float xx1 = fmaxf(x1, X1), yy1 = fmaxf(y1, Y1);
                float xx2 = fminf(x2, X2), yy2 = fminf(y2, Y2);
                float iw = fmaxf(xx2 - xx1 + 1.f, 0.f);
                float ih = fmaxf(yy2 - yy1 + 1.f, 0.f);
                float inter = iw * ih;
                float areaJ = (x2 - x1 + 1.f) * (y2 - y1 + 1.f);
                float iou = inter / (areaJ + areaI - inter);
                if (iou > NMS_TH) atomicAnd(&valid[j >> 5], ~(1u << (j & 31)));
            }
        }
        __syncthreads();
    }
    for (int i = tid; i < POST_NMS; i += 256) {
        int idx = keep[i];
        float* o = out + ((size_t)b * POST_NMS + i) * 5;
        o[0] = (float)b;
        if (idx >= 0) {
            o[1] = Bx[idx * 4 + 0]; o[2] = Bx[idx * 4 + 1];
            o[3] = Bx[idx * 4 + 2]; o[4] = Bx[idx * 4 + 3];
        } else {
            o[1] = 0.f; o[2] = 0.f; o[3] = 0.f; o[4] = 0.f;
        }
    }
}

extern "C" void kernel_launch(void* const* d_in, const int* in_sizes, int n_in,
                              void* d_out, int out_size, void* d_ws, size_t ws_size,
                              hipStream_t stream) {
    (void)in_sizes; (void)n_in; (void)out_size; (void)ws_size;
    const float* in      = (const float*)d_in[0];
    const float* im_info = (const float*)d_in[2];
    const float* Wf      = (const float*)d_in[3];
    const float* bf      = (const float*)d_in[4];
    const float* Wc      = (const float*)d_in[5];
    const float* bc      = (const float*)d_in[6];
    const float* Wr      = (const float*)d_in[7];
    const float* br      = (const float*)d_in[8];
    float* out = (float*)d_out;

    char* ws = (char*)d_ws;
    size_t off = 0;
    auto carve = [&](size_t bytes) -> void* {
        void* p = ws + off;
        off += (bytes + 255) & ~(size_t)255;
        return p;
    };
    _Float16* in16   = (_Float16*)carve((size_t)NPIX * CIN * sizeof(_Float16));
    _Float16* wfeat2 = (_Float16*)carve((size_t)KTOT * COUT * sizeof(_Float16));
    _Float16* wcat2  = (_Float16*)carve((size_t)CIN * NCAT * sizeof(_Float16));
    float*    bcat   = (float*)carve(NCAT * sizeof(float));
    _Float16* feat   = (_Float16*)carve((size_t)NPIX * COUT * sizeof(_Float16));
    float*    scores = (float*)carve((size_t)B_ * NANCH * sizeof(float));
    float*    boxes  = (float*)carve((size_t)B_ * NANCH * 4 * sizeof(float));
    float*    selS   = (float*)carve((size_t)B_ * PRE_NMS * sizeof(float));
    float*    selB   = (float*)carve((size_t)B_ * PRE_NMS * 4 * sizeof(float));

    pack_in16<<<((size_t)NPIX * CIN / 4 + 255) / 256, 256, 0, stream>>>(in, in16);
    pack_wfeat2<<<(KTOT * COUT + 255) / 256, 256, 0, stream>>>(Wf, wfeat2);
    pack_wcat2<<<(CIN * NCAT + 255) / 256, 256, 0, stream>>>(Wc, Wr, bc, br, wcat2, bcat);
    conv3x3_relu_wmma<<<dim3(NPIX / 128, COUT / 128), 256, 0, stream>>>(in16, wfeat2, bf, feat);
    head_decode_wmma<<<NPIX / 64, 256, 0, stream>>>(feat, wcat2, bcat, im_info, scores, boxes);
    topk_select<<<B_, 1024, 0, stream>>>(scores, boxes, selS, selB);
    nms_kernel<<<B_, 256, 0, stream>>>(selS, selB, out);
}
